// LanguageModel_12446815224552
// MI455X (gfx1250) — matrix operand order repaired and checked by reading, not compiled, not run on hardware
//
#include <hip/hip_runtime.h>
#include <cstdint>
#include <cmath>

// ---------------- model constants ----------------
constexpr int cB = 4, cL = 512, cV = 32000, cD = 1024, cH = 16, cNL = 2;
constexpr int cE = 8, cFF = 4096;
constexpr int cS = cB * cL;          // 2048 tokens
constexpr int cCAP = 320;            // int(1.25 * 2048 / 8)
constexpr int cHD = cD / cH;         // 64

typedef __bf16 bf16_t;
typedef bf16_t v16bf __attribute__((ext_vector_type(16)));
typedef float  v8f   __attribute__((ext_vector_type(8)));
typedef unsigned int v4u __attribute__((ext_vector_type(4)));

union Frag { v4u u[2]; v16bf v; };

static __device__ inline v4u v4u_zero() { v4u z; z.x = 0u; z.y = 0u; z.z = 0u; z.w = 0u; return z; }

// ---------------- generic WMMA GEMM:  C[z] = A[z] * B[z]^T * scale (+bias) ----------------
// A: M x K row-major bf16 (lda), B: N x K row-major bf16 (ldb), C: M x N (ldc) f32 or bf16.
// Block: 256 threads (8 wave32), tile 128x128, K-step 32. Wave = 32x64 (2x4 of 16x16 WMMA).
// Interior blocks take an unguarded fast path (12 b128 loads + 8 WMMA per K-step).
template <bool HAS_BIAS, bool RELU, bool OUT_BF16>
__global__ __launch_bounds__(256) void gemm_wmma_kernel(
    const bf16_t* __restrict__ A, const bf16_t* __restrict__ Bm,
    const float* __restrict__ bias, void* __restrict__ Cv,
    int M, int N, int Kd, long lda, long ldb, long ldc,
    long sA, long sB, long sC, long sBias, float scale)
{
    const int z = blockIdx.z;
    const bf16_t* Ap = A + (long)z * sA;
    const bf16_t* Bp = Bm + (long)z * sB;
    const float* biasp = HAS_BIAS ? (bias + (long)z * sBias) : nullptr;

    const int lane = threadIdx.x & 31;
    const int wid  = threadIdx.x >> 5;
    const int hl   = lane >> 4;      // lane half (0/1)
    const int l16  = lane & 15;
    const int m0 = blockIdx.y * 128 + (wid >> 1) * 32;  // wave M base
    const int n0 = blockIdx.x * 128 + (wid & 1) * 64;   // wave N base

    v8f acc[2][4];
#pragma unroll
    for (int i = 0; i < 2; ++i)
#pragma unroll
        for (int j = 0; j < 4; ++j)
#pragma unroll
            for (int e = 0; e < 8; ++e) acc[i][j][e] = 0.0f;

    const bool interior = ((int)blockIdx.y * 128 + 128 <= M) && ((int)blockIdx.x * 128 + 128 <= N);

    if (interior) {
        // precomputed per-fragment row base pointers; per-iteration math is just +k
        const bf16_t* pa[2];
        const bf16_t* pb[4];
#pragma unroll
        for (int ti = 0; ti < 2; ++ti) pa[ti] = Ap + (long)(m0 + ti * 16 + l16) * lda + 8 * hl;
#pragma unroll
        for (int tj = 0; tj < 4; ++tj) pb[tj] = Bp + (long)(n0 + tj * 16 + l16) * ldb + 8 * hl;

        for (int k = 0; k < Kd; k += 32) {
            Frag a[2], b[4];
#pragma unroll
            for (int ti = 0; ti < 2; ++ti) {
                a[ti].u[0] = *(const v4u*)(pa[ti] + k);
                a[ti].u[1] = *(const v4u*)(pa[ti] + k + 16);
                __builtin_prefetch(pa[ti] + k + 128, 0, 1);   // global_prefetch_b8, 256B ahead
            }
#pragma unroll
            for (int tj = 0; tj < 4; ++tj) {
                b[tj].u[0] = *(const v4u*)(pb[tj] + k);
                b[tj].u[1] = *(const v4u*)(pb[tj] + k + 16);
                __builtin_prefetch(pb[tj] + k + 128, 0, 1);
            }
#pragma unroll
            for (int ti = 0; ti < 2; ++ti)
#pragma unroll
                for (int tj = 0; tj < 4; ++tj)
                    acc[ti][tj] = __builtin_amdgcn_wmma_f32_16x16x32_bf16(
                        false, a[ti].v, false, b[tj].v, (short)0, acc[ti][tj], false, false);
        }
    } else {
        for (int k = 0; k < Kd; k += 32) {
            Frag a[2], b[4];
#pragma unroll
            for (int ti = 0; ti < 2; ++ti) {               // A frag: row per lane, K in 2 chunks
                int row = m0 + ti * 16 + l16;
                if (row < M) {
                    const bf16_t* p = Ap + (long)row * lda + k + 8 * hl;
                    a[ti].u[0] = *(const v4u*)p;
                    a[ti].u[1] = *(const v4u*)(p + 16);
                } else { a[ti].u[0] = v4u_zero(); a[ti].u[1] = v4u_zero(); }
            }
#pragma unroll
            for (int tj = 0; tj < 4; ++tj) {               // B frag: W^T row per lane, K in 2 chunks
                int rb = n0 + tj * 16 + l16;
                if (rb < N) {
                    const bf16_t* p = Bp + (long)rb * ldb + k + 8 * hl;
                    b[tj].u[0] = *(const v4u*)p;
                    b[tj].u[1] = *(const v4u*)(p + 16);
                } else { b[tj].u[0] = v4u_zero(); b[tj].u[1] = v4u_zero(); }
            }
#pragma unroll
            for (int ti = 0; ti < 2; ++ti)
#pragma unroll
                for (int tj = 0; tj < 4; ++tj)
                    acc[ti][tj] = __builtin_amdgcn_wmma_f32_16x16x32_bf16(
                        false, a[ti].v, false, b[tj].v, (short)0, acc[ti][tj], false, false);
        }
    }

    const long cz = (long)z * sC;
#pragma unroll
    for (int ti = 0; ti < 2; ++ti) {
#pragma unroll
        for (int v = 0; v < 8; ++v) {
            int row = m0 + ti * 16 + hl * 8 + v;
            if (row >= M) continue;
#pragma unroll
            for (int tj = 0; tj < 4; ++tj) {
                int col = n0 + tj * 16 + l16;
                if (col >= N) continue;
                float val = acc[ti][tj][v] * scale;
                if (HAS_BIAS) val += biasp[col];
                if (RELU) val = fmaxf(val, 0.0f);
                long off = cz + (long)row * ldc + col;
                if (OUT_BF16) ((bf16_t*)Cv)[off] = (bf16_t)val;
                else          ((float*)Cv)[off]  = val;
            }
        }
    }
}

// ---------------- wave32 block reductions (256 threads) ----------------
static __device__ inline float block_sum(float v, float* sh) {
    for (int off = 16; off; off >>= 1) v += __shfl_down(v, off, 32);
    int wid = threadIdx.x >> 5, lane = threadIdx.x & 31;
    if (lane == 0) sh[wid] = v;
    __syncthreads();
    if (wid == 0) {
        float t = (lane < 8) ? sh[lane] : 0.0f;
        for (int off = 4; off; off >>= 1) t += __shfl_down(t, off, 32);
        if (lane == 0) sh[8] = t;
    }
    __syncthreads();
    float r = sh[8];
    __syncthreads();
    return r;
}
static __device__ inline float block_max(float v, float* sh) {
    for (int off = 16; off; off >>= 1) v = fmaxf(v, __shfl_down(v, off, 32));
    int wid = threadIdx.x >> 5, lane = threadIdx.x & 31;
    if (lane == 0) sh[wid] = v;
    __syncthreads();
    if (wid == 0) {
        float t = (lane < 8) ? sh[lane] : -3.4e38f;
        for (int off = 4; off; off >>= 1) t = fmaxf(t, __shfl_down(t, off, 32));
        if (lane == 0) sh[8] = t;
    }
    __syncthreads();
    float r = sh[8];
    __syncthreads();
    return r;
}

// ---------------- embedding + sinusoidal PE ----------------
__global__ __launch_bounds__(256) void embed_pe_kernel(
    const int* __restrict__ src, const float* __restrict__ emb,
    float* __restrict__ xf, bf16_t* __restrict__ xb)
{
    const int s = blockIdx.x;
    const int pos = s % cL;
    const long tok = (long)src[s];
    const float c = -logf(10000.0f) / (float)cD;
#pragma unroll
    for (int i = 0; i < 4; ++i) {
        int d = threadIdx.x + 256 * i;
        int i2 = d & ~1;
        float arg = (float)pos * __expf((float)i2 * c);
        float pe = (d & 1) ? __cosf(arg) : __sinf(arg);
        float v = emb[tok * cD + d] * 32.0f + pe;   // sqrt(1024)=32
        xf[(long)s * cD + d] = v;
        xb[(long)s * cD + d] = (bf16_t)v;
    }
}

// ---------------- fp32 -> bf16 convert ----------------
__global__ __launch_bounds__(256) void cvt_f32_bf16(const float* __restrict__ in,
                                                    bf16_t* __restrict__ out, long n)
{
    long i = (long)blockIdx.x * blockDim.x + threadIdx.x;
    long stride = (long)gridDim.x * blockDim.x;
    for (; i < n; i += stride) out[i] = (bf16_t)in[i];
}

__global__ __launch_bounds__(256) void zero_f32(float* __restrict__ p, long n)
{
    long i = (long)blockIdx.x * blockDim.x + threadIdx.x;
    long stride = (long)gridDim.x * blockDim.x;
    for (; i < n; i += stride) p[i] = 0.0f;
}

// ---------------- causal softmax over one 512-wide row, writes bf16 probs ----------------
__global__ __launch_bounds__(256) void softmax_causal_kernel(
    const float* __restrict__ scores, bf16_t* __restrict__ attn)
{
    __shared__ float sh[9];
    const long r = blockIdx.x;           // [0, B*H*L)
    const int q = (int)(r % cL);
    const float* srow = scores + r * cL;
    bf16_t* arow = attn + r * cL;
    const int t = threadIdx.x;
    float v0 = (t       <= q) ? srow[t]       : -3.4e38f;
    float v1 = (t + 256 <= q) ? srow[t + 256] : -3.4e38f;
    float m = block_max(fmaxf(v0, v1), sh);
    float e0 = (t       <= q) ? __expf(v0 - m) : 0.0f;
    float e1 = (t + 256 <= q) ? __expf(v1 - m) : 0.0f;
    float s = block_sum(e0 + e1, sh);
    float inv = 1.0f / s;
    arow[t]       = (bf16_t)(e0 * inv);
    arow[t + 256] = (bf16_t)(e1 * inv);
}

// ---------------- transpose V heads: vT[b,h,d,l] = qkv[b,l, 2D + h*64 + d] ----------------
__global__ __launch_bounds__(256) void transpose_v_kernel(
    const bf16_t* __restrict__ qkv, bf16_t* __restrict__ vT)
{
    long idx = (long)blockIdx.x * 256 + threadIdx.x;   // total = B*H*64*L
    int l  = (int)(idx % cL);
    int dd = (int)((idx / cL) % cHD);
    int h  = (int)((idx / ((long)cL * cHD)) % cH);
    int b  = (int)(idx / ((long)cL * cHD * cH));
    vT[idx] = qkv[((long)(b * cL + l)) * (3 * cD) + 2 * cD + h * cHD + dd];
}

// ---------------- residual add + LayerNorm (writes f32 stream + bf16 copy) ----------------
__global__ __launch_bounds__(256) void residual_ln_kernel(
    const float* __restrict__ resid, const float* __restrict__ delta,
    const float* __restrict__ w, const float* __restrict__ bvec,
    float* __restrict__ outf, bf16_t* __restrict__ outb)
{
    __shared__ float sh[9];
    const long row = blockIdx.x;
    float vals[4];
    float s = 0.0f;
#pragma unroll
    for (int i = 0; i < 4; ++i) {
        int d = threadIdx.x + 256 * i;
        vals[i] = resid[row * cD + d] + delta[row * cD + d];
        s += vals[i];
    }
    float mean = block_sum(s, sh) * (1.0f / cD);
    float vs = 0.0f;
#pragma unroll
    for (int i = 0; i < 4; ++i) { float d = vals[i] - mean; vs += d * d; }
    float var = block_sum(vs, sh) * (1.0f / cD);
    float rstd = rsqrtf(var + 1e-5f);
#pragma unroll
    for (int i = 0; i < 4; ++i) {
        int d = threadIdx.x + 256 * i;
        float o = (vals[i] - mean) * rstd * w[d] + bvec[d];
        outf[row * cD + d] = o;
        outb[row * cD + d] = (bf16_t)o;
    }
}

// ---------------- MoE gate: logits, softmax, top-2, normalized weights ----------------
__global__ __launch_bounds__(256) void gate_topk_kernel(
    const float* __restrict__ x, const float* __restrict__ gw, const float* __restrict__ gb,
    int* __restrict__ topi, float* __restrict__ wnorm)
{
    __shared__ float lg[cE];
    const int s = blockIdx.x;
    const int wid = threadIdx.x >> 5, lane = threadIdx.x & 31;
    const float* xr = x + (long)s * cD;
    const float* wr = gw + (long)wid * cD;
    float p = 0.0f;
    for (int d = lane; d < cD; d += 32) p += xr[d] * wr[d];
    for (int off = 16; off; off >>= 1) p += __shfl_down(p, off, 32);
    if (lane == 0) lg[wid] = p + gb[wid];
    __syncthreads();
    if (threadIdx.x == 0) {
        float mx = lg[0];
        for (int e = 1; e < cE; ++e) mx = fmaxf(mx, lg[e]);
        float pr[cE]; float sum = 0.0f;
        for (int e = 0; e < cE; ++e) { pr[e] = __expf(lg[e] - mx); sum += pr[e]; }
        for (int e = 0; e < cE; ++e) pr[e] /= sum;
        int i0 = 0;
        for (int e = 1; e < cE; ++e) if (pr[e] > pr[i0]) i0 = e;
        int i1 = (i0 == 0) ? 1 : 0;
        for (int e = 0; e < cE; ++e) if (e != i0 && pr[e] > pr[i1]) i1 = e;
        float t = pr[i0] + pr[i1];
        topi[2 * s] = i0; topi[2 * s + 1] = i1;
        wnorm[2 * s] = pr[i0] / t; wnorm[2 * s + 1] = pr[i1] / t;
    }
}

// ---------------- capacity routing (stable token order, matches argsort semantics) ----------------
__global__ void route_kernel(const int* __restrict__ topi, const float* __restrict__ wnorm,
                             int* __restrict__ idx, float* __restrict__ wtok)
{
    const int e = blockIdx.x;
    if (threadIdx.x != 0) return;
    int cnt = 0;
    for (int s = 0; s < cS && cnt < cCAP; ++s) {
        int i0 = topi[2 * s], i1 = topi[2 * s + 1];
        if (i0 == e || i1 == e) {
            idx[e * cCAP + cnt] = s;
            wtok[e * cCAP + cnt] = (i0 == e ? wnorm[2 * s] : 0.0f) + (i1 == e ? wnorm[2 * s + 1] : 0.0f);
            ++cnt;
        }
    }
    for (; cnt < cCAP; ++cnt) { idx[e * cCAP + cnt] = -1; wtok[e * cCAP + cnt] = 0.0f; }
}

// ---------------- gather routed tokens (bf16) ----------------
__global__ __launch_bounds__(256) void gather_xe_kernel(
    const float* __restrict__ xf, const int* __restrict__ idx, bf16_t* __restrict__ xe)
{
    const int r = blockIdx.x;          // [0, E*CAP)
    const int s = idx[r];
#pragma unroll
    for (int i = 0; i < 4; ++i) {
        int d = threadIdx.x + 256 * i;
        xe[(long)r * cD + d] = (s >= 0) ? (bf16_t)xf[(long)s * cD + d] : (bf16_t)0.0f;
    }
}

// ---------------- weighted scatter-add of expert outputs ----------------
__global__ __launch_bounds__(256) void scatter_moe_kernel(
    const float* __restrict__ oe, const int* __restrict__ idx,
    const float* __restrict__ wtok, float* __restrict__ y)
{
    const int r = blockIdx.x;
    const int s = idx[r];
    if (s < 0) return;
    const float w = wtok[r];
#pragma unroll
    for (int i = 0; i < 4; ++i) {
        int d = threadIdx.x + 256 * i;
        atomicAdd(&y[(long)s * cD + d], oe[(long)r * cD + d] * w);
    }
}

// ---------------- host-side GEMM dispatch ----------------
static inline void gemm(const bf16_t* A, const bf16_t* B, const float* bias, void* C,
                        int M, int N, int K, long lda, long ldb, long ldc,
                        long sA, long sB, long sC, long sBias, int batch, float scale,
                        bool hasBias, bool relu, bool outBF16, hipStream_t stream)
{
    dim3 grid((N + 127) / 128, (M + 127) / 128, batch);
    dim3 block(256);
    if (hasBias && !relu && outBF16)
        gemm_wmma_kernel<true, false, true><<<grid, block, 0, stream>>>(A, B, bias, C, M, N, K, lda, ldb, ldc, sA, sB, sC, sBias, scale);
    else if (hasBias && relu && outBF16)
        gemm_wmma_kernel<true, true, true><<<grid, block, 0, stream>>>(A, B, bias, C, M, N, K, lda, ldb, ldc, sA, sB, sC, sBias, scale);
    else if (hasBias && !relu && !outBF16)
        gemm_wmma_kernel<true, false, false><<<grid, block, 0, stream>>>(A, B, bias, C, M, N, K, lda, ldb, ldc, sA, sB, sC, sBias, scale);
    else if (!hasBias && !relu && outBF16)
        gemm_wmma_kernel<false, false, true><<<grid, block, 0, stream>>>(A, B, bias, C, M, N, K, lda, ldb, ldc, sA, sB, sC, sBias, scale);
    else
        gemm_wmma_kernel<false, false, false><<<grid, block, 0, stream>>>(A, B, bias, C, M, N, K, lda, ldb, ldc, sA, sB, sC, sBias, scale);
}

extern "C" void kernel_launch(void* const* d_in, const int* in_sizes, int n_in,
                              void* d_out, int out_size, void* d_ws, size_t ws_size,
                              hipStream_t stream)
{
    const int*   src       = (const int*)  d_in[0];
    const float* emb       = (const float*)d_in[1];
    const float* in_proj_w = (const float*)d_in[2];
    const float* in_proj_b = (const float*)d_in[3];
    const float* out_proj_w= (const float*)d_in[4];
    const float* out_proj_b= (const float*)d_in[5];
    const float* ln1_w     = (const float*)d_in[6];
    const float* ln1_b     = (const float*)d_in[7];
    const float* ln2_w     = (const float*)d_in[8];
    const float* ln2_b     = (const float*)d_in[9];
    const float* gate_w    = (const float*)d_in[10];
    const float* gate_b    = (const float*)d_in[11];
    const float* w1        = (const float*)d_in[12];
    const float* b1        = (const float*)d_in[13];
    const float* w2        = (const float*)d_in[14];
    const float* b2        = (const float*)d_in[15];
    const float* dec_w     = (const float*)d_in[16];
    const float* dec_bias  = (const float*)d_in[17];
    float* out = (float*)d_out;

    // workspace carve-out (256B aligned)
    char* ws = (char*)d_ws;
    auto alloc = [&](size_t bytes) -> void* {
        void* p = (void*)ws; ws += (bytes + 255) & ~(size_t)255; return p;
    };
    float*  xf      = (float*) alloc((size_t)cS * cD * 4);
    bf16_t* xb      = (bf16_t*)alloc((size_t)cS * cD * 2);
    bf16_t* qkv     = (bf16_t*)alloc((size_t)cS * 3 * cD * 2);
    float*  scores  = (float*) alloc((size_t)cB * cH * cL * cL * 4);
    bf16_t* attn    = (bf16_t*)alloc((size_t)cB * cH * cL * cL * 2);
    bf16_t* vT      = (bf16_t*)alloc((size_t)cS * cD * 2);
    bf16_t* attnout = (bf16_t*)alloc((size_t)cS * cD * 2);
    float*  proj    = (float*) alloc((size_t)cS * cD * 4);
    bf16_t* wqkv_b  = (bf16_t*)alloc((size_t)3 * cD * cD * 2);
    bf16_t* wo_b    = (bf16_t*)alloc((size_t)cD * cD * 2);
    bf16_t* wbig1   = (bf16_t*)alloc((size_t)cE * cFF * cD * 2);   // w1[layer] bf16, reused for dec_w
    bf16_t* wbig2   = (bf16_t*)alloc((size_t)cE * cD * cFF * 2);   // w2[layer] bf16
    int*    topi    = (int*)   alloc((size_t)cS * 2 * 4);
    float*  wnorm   = (float*) alloc((size_t)cS * 2 * 4);
    int*    ridx    = (int*)   alloc((size_t)cE * cCAP * 4);
    float*  wtok    = (float*) alloc((size_t)cE * cCAP * 4);
    bf16_t* xe      = (bf16_t*)alloc((size_t)cE * cCAP * cD * 2);
    bf16_t* hbuf    = (bf16_t*)alloc((size_t)cE * cCAP * cFF * 2);
    float*  oe      = (float*) alloc((size_t)cE * cCAP * cD * 4);
    float*  ybuf    = (float*) alloc((size_t)cS * cD * 4);

    // 1) embedding + positional encoding
    embed_pe_kernel<<<cS, 256, 0, stream>>>(src, emb, xf, xb);

    for (int layer = 0; layer < cNL; ++layer) {
        const float* wi = in_proj_w + (long)layer * 3 * cD * cD;
        const float* bi = in_proj_b + (long)layer * 3 * cD;
        const float* wo = out_proj_w + (long)layer * cD * cD;
        const float* bo = out_proj_b + (long)layer * cD;
        const float* l1w = ln1_w + (long)layer * cD;
        const float* l1b = ln1_b + (long)layer * cD;
        const float* l2w = ln2_w + (long)layer * cD;
        const float* l2b = ln2_b + (long)layer * cD;
        const float* gw = gate_w + (long)layer * cE * cD;
        const float* gb = gate_b + (long)layer * cE;
        const float* w1l = w1 + (long)layer * cE * cFF * cD;
        const float* b1l = b1 + (long)layer * cE * cFF;
        const float* w2l = w2 + (long)layer * cE * cD * cFF;
        const float* b2l = b2 + (long)layer * cE * cD;

        // weight conversion to bf16
        cvt_f32_bf16<<<4096, 256, 0, stream>>>(wi,  wqkv_b, (long)3 * cD * cD);
        cvt_f32_bf16<<<4096, 256, 0, stream>>>(wo,  wo_b,   (long)cD * cD);
        cvt_f32_bf16<<<8192, 256, 0, stream>>>(w1l, wbig1,  (long)cE * cFF * cD);
        cvt_f32_bf16<<<8192, 256, 0, stream>>>(w2l, wbig2,  (long)cE * cD * cFF);

        // qkv = x @ wi^T + bi  (bf16 out)
        gemm(xb, wqkv_b, bi, qkv, cS, 3 * cD, cD, cD, cD, 3 * cD,
             0, 0, 0, 0, 1, 1.0f, true, false, true, stream);

        // scores[b,h] = Q K^T / 8   (per-b launch, batch over heads via strides)
        for (int b = 0; b < cB; ++b) {
            const bf16_t* Qb = qkv + (long)b * cL * 3 * cD;
            const bf16_t* Kb = Qb + cD;
            float* Cb = scores + (long)b * cH * cL * cL;
            gemm(Qb, Kb, nullptr, Cb, cL, cL, cHD, 3 * cD, 3 * cD, cL,
                 cHD, cHD, (long)cL * cL, 0, cH, 0.125f, false, false, false, stream);
        }

        // causal softmax -> bf16 attn
        softmax_causal_kernel<<<cB * cH * cL, 256, 0, stream>>>(scores, attn);

        // transpose V per head
        transpose_v_kernel<<<(cS * cD) / 256, 256, 0, stream>>>(qkv, vT);

        // attnout[b, :, h*64:(h+1)*64] = attn[b,h] @ V[b,h]  (bf16 out)
        for (int b = 0; b < cB; ++b) {
            const bf16_t* Ab = attn + (long)b * cH * cL * cL;
            const bf16_t* Bb = vT + (long)b * cH * cHD * cL;
            bf16_t* Cb = attnout + (long)b * cL * cD;
            gemm(Ab, Bb, nullptr, Cb, cL, cHD, cL, cL, cL, cD,
                 (long)cL * cL, (long)cHD * cL, cHD, 0, cH, 1.0f, false, false, true, stream);
        }

        // out-proj (f32 out) then residual + LN1
        gemm(attnout, wo_b, bo, proj, cS, cD, cD, cD, cD, cD,
             0, 0, 0, 0, 1, 1.0f, true, false, false, stream);
        residual_ln_kernel<<<cS, 256, 0, stream>>>(xf, proj, l1w, l1b, xf, xb);

        // ---- MoE ----
        gate_topk_kernel<<<cS, 256, 0, stream>>>(xf, gw, gb, topi, wnorm);
        route_kernel<<<cE, 32, 0, stream>>>(topi, wnorm, ridx, wtok);
        gather_xe_kernel<<<cE * cCAP, 256, 0, stream>>>(xf, ridx, xe);

        // h = relu(xe @ w1^T + b1)  (per-expert batch, bf16 out)
        gemm(xe, wbig1, b1l, hbuf, cCAP, cFF, cD, cD, cD, cFF,
             (long)cCAP * cD, (long)cFF * cD, (long)cCAP * cFF, cFF, cE,
             1.0f, true, true, true, stream);
        // oe = h @ w2^T + b2  (f32 out)
        gemm(hbuf, wbig2, b2l, oe, cCAP, cD, cFF, cFF, cFF, cD,
             (long)cCAP * cFF, (long)cD * cFF, (long)cCAP * cD, cD, cE,
             1.0f, true, false, false, stream);

        zero_f32<<<4096, 256, 0, stream>>>(ybuf, (long)cS * cD);
        scatter_moe_kernel<<<cE * cCAP, 256, 0, stream>>>(oe, ridx, wtok, ybuf);

        residual_ln_kernel<<<cS, 256, 0, stream>>>(xf, ybuf, l2w, l2b, xf, xb);
    }

    // 3) decode: logits = x @ dec_w^T + dec_b  (reuse wbig1 for bf16 dec_w)
    cvt_f32_bf16<<<8192, 256, 0, stream>>>(dec_w, wbig1, (long)cV * cD);
    gemm(xb, wbig1, dec_bias, out, cS, cV, cD, cD, cD, cV,
         0, 0, 0, 0, 1, 1.0f, true, false, false, stream);
}
